// MultiHeadAttention_71622874628693
// MI455X (gfx1250) — matrix operand order repaired and checked by reading, not compiled, not run on hardware
//
#include <hip/hip_runtime.h>
#include <math.h>

typedef __attribute__((ext_vector_type(16))) __bf16 v16bf;
typedef __attribute__((ext_vector_type(8)))  float  v8f;

#define D_MODEL 1024
#define N_HEADS 16
#define D_HEAD  64
#define BATCH   4
#define SEQ     2048

union Frag {            // one WMMA 16-bit A/B operand (32 B per lane)
  v16bf v;
  uint4 q[2];
  unsigned short u[16];
};

__device__ __forceinline__ unsigned short f32_bf16(float f) {
  unsigned int u = __float_as_uint(f);
  u += 0x7FFFu + ((u >> 16) & 1u);           // round-to-nearest-even
  return (unsigned short)(u >> 16);
}
__device__ __forceinline__ v8f v8f_zero() {
  v8f z = {0.f,0.f,0.f,0.f,0.f,0.f,0.f,0.f}; return z;
}
__device__ __forceinline__ float rmax16(float v) {
#pragma unroll
  for (int m = 1; m < 16; m <<= 1) v = fmaxf(v, __shfl_xor(v, m, 32));
  return v;
}
__device__ __forceinline__ float rsum16(float v) {
#pragma unroll
  for (int m = 1; m < 16; m <<= 1) v += __shfl_xor(v, m, 32);
  return v;
}

// fp32 -> bf16 bulk conversion (4 elems/thread)
__global__ __launch_bounds__(256) void pack_bf16(
    const float* __restrict__ src, unsigned short* __restrict__ dst, int n4) {
  int i = blockIdx.x * 256 + threadIdx.x;
  if (i >= n4) return;
  float4 v = ((const float4*)src)[i];
  ushort4 o;
  o.x = f32_bf16(v.x); o.y = f32_bf16(v.y);
  o.z = f32_bf16(v.z); o.w = f32_bf16(v.w);
  ((ushort4*)dst)[i] = o;
}

// C[M,N] = A[M,K] @ B[K,N], bf16 in, bf16 or fp32 out.
// 256 thr (8 waves), block tile 128x128, k-step 32, wave tile 32x64.
// All fragment loads are 2x ds_load_b128 (A row-major LDS, B transposed LDS).
#define BM 128
#define BN 128
#define BK 32
#define LDSS 48   // padded LDS row stride (ushorts): 96 B, keeps b128 alignment

__global__ __launch_bounds__(256) void gemm_bf16_wmma(
    const unsigned short* __restrict__ A, const unsigned short* __restrict__ B,
    float* __restrict__ Cf, unsigned short* __restrict__ Cb,
    int M, int N, int K) {
  __shared__ __align__(16) unsigned short sA[BM * LDSS];   // 12 KB
  __shared__ __align__(16) unsigned short sBt[BN * LDSS];  // 12 KB
  const int tid = threadIdx.x, lane = tid & 31, w = tid >> 5;
  const int half = lane >> 4, n16 = lane & 15;
  const int wm = (w & 3) * 32, wn = (w >> 2) * 64;
  const int bm = blockIdx.y * BM, bn = blockIdx.x * BN;

  v8f acc[2][4];
#pragma unroll
  for (int mt = 0; mt < 2; ++mt)
#pragma unroll
    for (int nt = 0; nt < 4; ++nt) acc[mt][nt] = v8f_zero();

  const int arow = tid >> 1, acol = (tid & 1) * 16;   // A stage: 16 bf16/thread
  const int bk   = tid >> 3, bnc  = (tid & 7) * 16;   // B stage: 16 bf16/thread

  for (int k0 = 0; k0 < K; k0 += BK) {
    {  // A tile 128x32 -> row-major LDS (2x b128 in, 2x b128 out)
      const uint4* src = (const uint4*)(A + (size_t)(bm + arow) * K + k0 + acol);
      uint4 a0 = src[0], a1 = src[1];
      *(uint4*)(sA + arow * LDSS + acol)     = a0;
      *(uint4*)(sA + arow * LDSS + acol + 8) = a1;
    }
    {  // B tile 32x128 -> transposed LDS sBt[n][k]
      union { uint4 q[2]; unsigned short u[16]; } t;
      const uint4* src = (const uint4*)(B + (size_t)(k0 + bk) * N + bn + bnc);
      t.q[0] = src[0]; t.q[1] = src[1];
#pragma unroll
      for (int e = 0; e < 16; ++e)
        sBt[(bnc + e) * LDSS + bk] = t.u[e];
    }
    if (k0 + BK < K)  // global_prefetch_b8 of next A tile
      __builtin_prefetch(A + (size_t)(bm + arow) * K + k0 + BK + acol, 0, 0);
    __syncthreads();

    Frag af[2], bfr[4];
#pragma unroll
    for (int mt = 0; mt < 2; ++mt) {   // A-frag: two contiguous 8-elem K-runs
      const unsigned short* p = sA + (wm + 16 * mt + n16) * LDSS + 8 * half;
      af[mt].q[0] = *(const uint4*)p;
      af[mt].q[1] = *(const uint4*)(p + 16);
    }
#pragma unroll
    for (int nt = 0; nt < 4; ++nt) {   // B-frag: two contiguous 8-elem K-runs
      const unsigned short* p = sBt + (wn + 16 * nt + n16) * LDSS + 8 * half;
      bfr[nt].q[0] = *(const uint4*)p;
      bfr[nt].q[1] = *(const uint4*)(p + 16);
    }
#pragma unroll
    for (int mt = 0; mt < 2; ++mt)
#pragma unroll
      for (int nt = 0; nt < 4; ++nt)
        acc[mt][nt] = __builtin_amdgcn_wmma_f32_16x16x32_bf16(
            false, af[mt].v, false, bfr[nt].v, (short)0, acc[mt][nt], false, false);
    __syncthreads();
  }

#pragma unroll
  for (int mt = 0; mt < 2; ++mt)
#pragma unroll
    for (int nt = 0; nt < 4; ++nt)
#pragma unroll
      for (int r = 0; r < 8; ++r) {
        int m = bm + wm + 16 * mt + r + 8 * half;
        int n = bn + wn + 16 * nt + n16;
        if (Cf) Cf[(size_t)m * N + n] = acc[mt][nt][r];
        else    Cb[(size_t)m * N + n] = f32_bf16(acc[mt][nt][r]);
      }
}

// Flash attention, bf16 QKV in / bf16 attn-out. One wave per 16-query tile.
// qkv row layout: [sel*1024 + h*64 + dh], row stride 3072, padded +32 rows.
__global__ __launch_bounds__(32) void flash_attn_wmma(
    const unsigned short* __restrict__ qkv, unsigned short* __restrict__ attn_out) {
  __shared__ __align__(16) unsigned short sP[16 * 32];
  const int lane = threadIdx.x;
  const int qb = blockIdx.x * 16;
  const int h = blockIdx.y, b = blockIdx.z;
  const int half = lane >> 4, n16 = lane & 15;
  const size_t row3 = 3 * D_MODEL;
  const size_t base = (size_t)b * SEQ;
  const float scale = 0.125f;                // 1/sqrt(64)

  Frag aq[2];                                // Q: 2x (2x global b128)
#pragma unroll
  for (int s = 0; s < 2; ++s) {
    const unsigned short* p =
        qkv + (base + qb + n16) * row3 + h * 64 + 32 * s + 8 * half;
    aq[s].q[0] = *(const uint4*)p;
    aq[s].q[1] = *(const uint4*)(p + 16);
  }

  float mrow[8], lrow[8], factor[8];
  v8f accO[4];
#pragma unroll
  for (int r = 0; r < 8; ++r) { mrow[r] = -__builtin_inff(); lrow[r] = 0.f; }
#pragma unroll
  for (int d = 0; d < 4; ++d) accO[d] = v8f_zero();

  const int nch = (qb + 16 + 31) >> 5;       // 32 keys / iteration
  for (int c = 0; c < nch; ++c) {
    const int kb = c * 32;
    v8f S[2];
#pragma unroll
    for (int t = 0; t < 2; ++t) {            // S = Q @ K^T  (4 WMMAs)
      const unsigned short* kp =
          qkv + (base + kb + 16 * t + n16) * row3 + D_MODEL + h * 64 + 8 * half;
      v8f sacc = v8f_zero();
#pragma unroll
      for (int s = 0; s < 2; ++s) {
        Frag bk;
        bk.q[0] = *(const uint4*)(kp + 32 * s);
        bk.q[1] = *(const uint4*)(kp + 32 * s + 16);
        sacc = __builtin_amdgcn_wmma_f32_16x16x32_bf16(
            false, aq[s].v, false, bk.v, (short)0, sacc, false, false);
      }
      S[t] = sacc;
    }
    // causal mask + online softmax (row stats live in C-layout halves)
#pragma unroll
    for (int r = 0; r < 8; ++r) {
      int q = qb + r + 8 * half;
      float s0 = ((kb + n16)      <= q) ? scale * S[0][r] : -__builtin_inff();
      float s1 = ((kb + 16 + n16) <= q) ? scale * S[1][r] : -__builtin_inff();
      float mnew = fmaxf(mrow[r], rmax16(fmaxf(s0, s1)));
      float p0 = __expf(s0 - mnew);
      float p1 = __expf(s1 - mnew);
      float f  = __expf(mrow[r] - mnew);
      lrow[r] = lrow[r] * f + rsum16(p0 + p1);
      mrow[r] = mnew;
      factor[r] = f;
      int m = r + 8 * half;
      sP[m * 32 + n16]      = f32_bf16(p0);  // C-layout -> row-major LDS
      sP[m * 32 + 16 + n16] = f32_bf16(p1);
    }
    asm volatile("s_wait_dscnt 0x0" ::: "memory");  // cross-lane LDS RAW

    Frag pf;                                 // P as A-frag: 2x ds_load_b128
    {
      const unsigned short* pp = sP + n16 * 32 + 16 * half;
      pf.q[0] = *(const uint4*)pp;
      pf.q[1] = *(const uint4*)(pp + 8);
    }
#pragma unroll
    for (int dt = 0; dt < 4; ++dt) {         // O += P @ V  (4 WMMAs)
      const unsigned short* vp = qkv + (base + kb + 16 * half) * row3 +
                                 2 * D_MODEL + h * 64 + dt * 16 + n16;
      Frag bv;                               // immediate-offset u16 gathers
#pragma unroll
      for (int i = 0; i < 16; ++i) bv.u[i] = vp[(size_t)i * row3];
      v8f o = accO[dt];
#pragma unroll
      for (int r = 0; r < 8; ++r) o[r] *= factor[r];
      accO[dt] = __builtin_amdgcn_wmma_f32_16x16x32_bf16(
          false, pf.v, false, bv.v, (short)0, o, false, false);
    }
  }
#pragma unroll
  for (int dt = 0; dt < 4; ++dt)
#pragma unroll
    for (int r = 0; r < 8; ++r) {
      int m = r + 8 * half;
      int d = dt * 16 + n16;
      attn_out[(base + qb + m) * (size_t)D_MODEL + h * 64 + d] =
          f32_bf16(accO[dt][r] / lrow[r]);
    }
}

extern "C" void kernel_launch(void* const* d_in, const int* in_sizes, int n_in,
                              void* d_out, int out_size, void* d_ws, size_t ws_size,
                              hipStream_t stream) {
  (void)in_sizes; (void)n_in; (void)out_size; (void)ws_size;
  const float* x     = (const float*)d_in[0];
  const float* w_qkv = (const float*)d_in[1];
  const float* w_out = (const float*)d_in[2];
  float* out = (float*)d_out;

  const int M = BATCH * SEQ;                               // 8192
  unsigned short* ws    = (unsigned short*)d_ws;
  unsigned short* Xb    = ws;                              size_t o = (size_t)M * D_MODEL;
  unsigned short* Wqkvb = ws + o;                          o += (size_t)D_MODEL * 3 * D_MODEL;
  unsigned short* Woutb = ws + o;                          o += (size_t)D_MODEL * D_MODEL;
  unsigned short* qkvb  = ws + o;                          o += (size_t)(M + 32) * 3 * D_MODEL; // +32 pad rows
  unsigned short* attnb = ws + o;

  pack_bf16<<<(M * D_MODEL / 4 + 255) / 256, 256, 0, stream>>>(x, Xb, M * D_MODEL / 4);
  pack_bf16<<<(3 * D_MODEL * D_MODEL / 4 + 255) / 256, 256, 0, stream>>>(
      w_qkv, Wqkvb, 3 * D_MODEL * D_MODEL / 4);
  pack_bf16<<<(D_MODEL * D_MODEL / 4 + 255) / 256, 256, 0, stream>>>(
      w_out, Woutb, D_MODEL * D_MODEL / 4);

  gemm_bf16_wmma<<<dim3(3 * D_MODEL / BN, M / BM), 256, 0, stream>>>(
      Xb, Wqkvb, nullptr, qkvb, M, 3 * D_MODEL, D_MODEL);
  flash_attn_wmma<<<dim3(SEQ / 16, N_HEADS, BATCH), 32, 0, stream>>>(qkvb, attnb);
  gemm_bf16_wmma<<<dim3(D_MODEL / BN, M / BM), 256, 0, stream>>>(
      attnb, Woutb, out, nullptr, M, D_MODEL, D_MODEL);
}
